// LSTM_28604482191842
// MI455X (gfx1250) — matrix operand order repaired and checked by reading, not compiled, not run on hardware
//
#include <hip/hip_runtime.h>

#define H 4096
#define T 50
#define E 50

typedef __attribute__((ext_vector_type(16))) __bf16 v16bf;
typedef __attribute__((ext_vector_type(8)))  float  v8f;

union Frag { v16bf v; uint4 u[2]; };

// ---------------------------------------------------------------------------
// Prep: embedding gather -> ws_x[50*50], conv/linear chain -> scalar r (ws_r)
// ---------------------------------------------------------------------------
__global__ __launch_bounds__(128)
void prep_kernel(const int* __restrict__ inp, const float* __restrict__ emb,
                 const float* __restrict__ c1w, const float* __restrict__ c1b,
                 const float* __restrict__ c2w, const float* __restrict__ c2b,
                 const float* __restrict__ l1w, const float* __restrict__ l1b,
                 const float* __restrict__ l2w, const float* __restrict__ l2b,
                 const float* __restrict__ l3w, const float* __restrict__ l3b,
                 float* __restrict__ wsX, float* __restrict__ wsR)
{
    __shared__ float sx[T * E];
    __shared__ float h1[62];
    __shared__ float h2[52];
    __shared__ float f1[35];
    __shared__ float f2[15];
    const int tid = threadIdx.x;

    for (int i = tid; i < T * E; i += 128) {
        int t = i / E, e = i % E;
        float v = emb[(size_t)inp[t] * E + e];
        sx[i] = v;
        wsX[i] = v;
    }
    __syncthreads();

    // conv1: kernel (50,9) over zero-padded (50,70) -> 62 outputs, relu
    if (tid < 62) {
        float s = c1b[0];
        for (int a = 0; a < 50; ++a)
            for (int bb = 0; bb < 9; ++bb) {
                int src = tid + bb - 10;                 // padded col - 10
                if (src >= 0 && src < 50) s += sx[a * E + src] * c1w[a * 9 + bb];
            }
        h1[tid] = s > 0.f ? s : 0.f;
    }
    __syncthreads();
    // conv2: kernel (1,11) -> 52 outputs, relu
    if (tid < 52) {
        float s = c2b[0];
        for (int bb = 0; bb < 11; ++bb) s += h1[tid + bb] * c2w[bb];
        h2[tid] = s > 0.f ? s : 0.f;
    }
    __syncthreads();
    if (tid < 35) {
        float s = l1b[tid];
        for (int k = 0; k < 52; ++k) s += l1w[tid * 52 + k] * h2[k];
        f1[tid] = s > 0.f ? s : 0.f;
    }
    __syncthreads();
    if (tid < 15) {
        float s = l2b[tid];
        for (int k = 0; k < 35; ++k) s += l2w[tid * 35 + k] * f1[k];
        f2[tid] = s > 0.f ? s : 0.f;
    }
    __syncthreads();
    if (tid == 0) {
        float s = l3b[0];
        for (int k = 0; k < 15; ++k) s += l3w[k] * f2[k];
        wsR[0] = s;
    }
}

// ---------------------------------------------------------------------------
// Pack dale_hh = relu(W_hh) * diag(sign) into bf16 A-fragments.
// Layout: chunk (b, c) = rows [16b,16b+16) x K [32c,32c+32); per chunk 32
// lanes x 16 bf16 (32B contiguous per lane) matching V_WMMA A layout:
//   lane: M = lane&15, hi = lane>>4; VGPR v pair K = (v<4?0:16)+hi*8+2*(v&3)
// ---------------------------------------------------------------------------
__global__ __launch_bounds__(256)
void pack_kernel(const float* __restrict__ whh, const float* __restrict__ dsg,
                 unsigned short* __restrict__ wpk)
{
    const int tid  = blockIdx.x * 256 + threadIdx.x;   // 0 .. 256*128*32-1
    const int lane = tid & 31;
    const int c    = (tid >> 5) & 127;
    const int b    = tid >> 12;
    const int m    = lane & 15;
    const int hi   = lane >> 4;
    const int row  = b * 16 + m;
    const float* wr = whh + (size_t)row * H;

    Frag a;
#pragma unroll
    for (int v = 0; v < 8; ++v) {
        int k = c * 32 + ((v < 4) ? 0 : 16) + hi * 8 + (v & 3) * 2;
        float e0 = wr[k];     e0 = e0 > 0.f ? e0 * dsg[k]     : 0.f;
        float e1 = wr[k + 1]; e1 = e1 > 0.f ? e1 * dsg[k + 1] : 0.f;
        a.v[2 * v]     = (__bf16)e0;
        a.v[2 * v + 1] = (__bf16)e1;
    }
    uint4* dst = (uint4*)((char*)wpk + (size_t)tid * 32);
    dst[0] = a.u[0];
    dst[1] = a.u[1];
}

// ---------------------------------------------------------------------------
// One scan step. 128 blocks x 256 threads (8 waves). Block owns 32 rows
// (two 16-row WMMA blocks); each row block is split 4-ways over K across
// waves, partials reduced through LDS. h broadcast across B columns.
// ---------------------------------------------------------------------------
template <bool PACKED>
__global__ __launch_bounds__(256)
void step_kernel(int t,
                 const float* __restrict__ wsX, const float* __restrict__ wsR,
                 const float* __restrict__ wih, const float* __restrict__ bih,
                 const float* __restrict__ bhh,
                 const float* __restrict__ whh, const float* __restrict__ dsg,
                 const unsigned short* __restrict__ wpk,
                 float* __restrict__ hid)          // = d_out + 2, [T][H]
{
    __shared__ float sh_xt[64];
    __shared__ float sh_pre[32];
    __shared__ float part[8][16];
    __shared__ __align__(16) __bf16 sh_hb[H];

    const int tid  = threadIdx.x;
    const int lane = tid & 31;
    const int w    = tid >> 5;
    const int rowBlock0 = blockIdx.x * 32;
    const float* hprev = hid + (size_t)(t - 1) * H;

    // stage x_t and h_{t-1} (bf16)
    if (tid < E) sh_xt[tid] = wsX[t * E + tid];
    if (t > 0) {
#pragma unroll
        for (int i = 0; i < 16; ++i) {
            int j = tid * 16 + i;
            sh_hb[j] = (__bf16)hprev[j];
        }
    }
    __syncthreads();

    // pre[row] = bias_ih + bias_hh + x_t . W_ih[row]
    if (tid < 32) {
        int row = rowBlock0 + tid;
        float s = bih[row] + bhh[row];
        const float* wr = wih + (size_t)row * E;
#pragma unroll 10
        for (int e = 0; e < E; ++e) s += sh_xt[e] * wr[e];
        sh_pre[tid] = s;
    }

    const int b  = blockIdx.x * 2 + (w >> 2);   // global 16-row block
    const int ks = w & 3;                       // K slice (quarter of 128 chunks)
    const int hi = lane >> 4;

    v8f acc = {};
    if (t > 0) {
        if (PACKED) {
            const char* base = (const char*)wpk + (size_t)b * 128 * 1024 + (size_t)lane * 32;
            for (int c = ks * 32; c < ks * 32 + 32; ++c) {
                Frag a;
                const uint4* p = (const uint4*)(base + (size_t)c * 1024);
                a.u[0] = p[0];
                a.u[1] = p[1];
                Frag bf;
                const uint4* q = (const uint4*)&sh_hb[c * 32 + hi * 8];
                bf.u[0] = q[0];
                bf.u[1] = q[2];
                acc = __builtin_amdgcn_wmma_f32_16x16x32_bf16(
                        false, a.v, false, bf.v, (short)0, acc, false, false);
            }
        } else {
            const int row = b * 16 + (lane & 15);
            const float* wr = whh + (size_t)row * H;
            for (int c = ks * 32; c < ks * 32 + 32; ++c) {
                Frag a;
#pragma unroll
                for (int v = 0; v < 8; ++v) {
                    int k = c * 32 + ((v < 4) ? 0 : 16) + hi * 8 + (v & 3) * 2;
                    float e0 = wr[k];     e0 = e0 > 0.f ? e0 * dsg[k]     : 0.f;
                    float e1 = wr[k + 1]; e1 = e1 > 0.f ? e1 * dsg[k + 1] : 0.f;
                    a.v[2 * v]     = (__bf16)e0;
                    a.v[2 * v + 1] = (__bf16)e1;
                }
                Frag bf;
                const uint4* q = (const uint4*)&sh_hb[c * 32 + hi * 8];
                bf.u[0] = q[0];
                bf.u[1] = q[2];
                acc = __builtin_amdgcn_wmma_f32_16x16x32_bf16(
                        false, a.v, false, bf.v, (short)0, acc, false, false);
            }
        }
    }

    // column N=0 of D lives in lanes 0 (M=0..7) and 16 (M=8..15)
    if (lane == 0 || lane == 16) {
        const int mofs = (lane == 16) ? 8 : 0;
#pragma unroll
        for (int i = 0; i < 8; ++i) part[w][mofs + i] = acc[i];
    }
    __syncthreads();

    if (tid < 32) {
        const float r = wsR[0];
        const int m  = tid & 15;
        const int rb = tid >> 4;
        float mv = part[rb * 4 + 0][m] + part[rb * 4 + 1][m]
                 + part[rb * 4 + 2][m] + part[rb * 4 + 3][m];
        int grow = rowBlock0 + rb * 16 + m;
        float pre  = sh_pre[tid] + mv;
        float hold = (t > 0) ? hprev[grow] : 0.f;
        float hn   = r * hold + (1.f - r) * pre;
        hid[(size_t)t * H + grow] = hn > 0.f ? hn : 0.f;
    }
}

// ---------------------------------------------------------------------------
// all_outputs[t] = hid[t] . out_w[o] + out_b[o]; softmax_out == row t=49
// ---------------------------------------------------------------------------
__global__ __launch_bounds__(256)
void out_kernel(const float* __restrict__ hid, const float* __restrict__ ow,
                const float* __restrict__ ob, float* __restrict__ out)
{
    __shared__ float red[256];
    const int t = blockIdx.x >> 1;
    const int o = blockIdx.x & 1;
    const int tid = threadIdx.x;
    const float* h = hid + (size_t)t * H;
    const float* w = ow + (size_t)o * H;
    float s = 0.f;
    for (int j = tid; j < H; j += 256) s += h[j] * w[j];
    red[tid] = s;
    __syncthreads();
    for (int st = 128; st > 0; st >>= 1) {
        if (tid < st) red[tid] += red[tid + st];
        __syncthreads();
    }
    if (tid == 0) {
        float v = red[0] + ob[o];
        out[2 + T * H + blockIdx.x] = v;   // all_outputs
        if (t == T - 1) out[o] = v;        // softmax_out
    }
}

// ---------------------------------------------------------------------------
extern "C" void kernel_launch(void* const* d_in, const int* in_sizes, int n_in,
                              void* d_out, int out_size, void* d_ws, size_t ws_size,
                              hipStream_t stream)
{
    (void)in_sizes; (void)n_in; (void)out_size;
    const int*   inp = (const int*)  d_in[0];
    const float* emb = (const float*)d_in[1];
    const float* c1w = (const float*)d_in[2];
    const float* c1b = (const float*)d_in[3];
    const float* c2w = (const float*)d_in[4];
    const float* c2b = (const float*)d_in[5];
    const float* l1w = (const float*)d_in[6];
    const float* l1b = (const float*)d_in[7];
    const float* l2w = (const float*)d_in[8];
    const float* l2b = (const float*)d_in[9];
    const float* l3w = (const float*)d_in[10];
    const float* l3b = (const float*)d_in[11];
    const float* wih = (const float*)d_in[12];
    const float* whh = (const float*)d_in[13];
    const float* bih = (const float*)d_in[14];
    const float* bhh = (const float*)d_in[15];
    const float* dsg = (const float*)d_in[16];
    const float* ow  = (const float*)d_in[17];
    const float* ob  = (const float*)d_in[18];

    float* out = (float*)d_out;
    float* hid = out + 2;                      // all_hidden lives in d_out
    float* wsX = (float*)d_ws;                 // [2500]
    float* wsR = wsX + 2500;                   // [1]
    unsigned short* wpk = (unsigned short*)((char*)d_ws + 16384);
    const size_t need = 16384 + (size_t)H * H * 2;   // ~33.6 MB bf16 pack
    const bool packed = ws_size >= need;

    prep_kernel<<<1, 128, 0, stream>>>(inp, emb, c1w, c1b, c2w, c2b,
                                       l1w, l1b, l2w, l2b, l3w, l3b, wsX, wsR);
    if (packed)
        pack_kernel<<<4096, 256, 0, stream>>>(whh, dsg, wpk);

    for (int t = 0; t < T; ++t) {
        if (packed)
            step_kernel<true><<<128, 256, 0, stream>>>(t, wsX, wsR, wih, bih, bhh,
                                                       whh, dsg, wpk, hid);
        else
            step_kernel<false><<<128, 256, 0, stream>>>(t, wsX, wsR, wih, bih, bhh,
                                                        whh, dsg, wpk, hid);
    }
    out_kernel<<<100, 256, 0, stream>>>(hid, ow, ob, out);
}
